// MPSClassifier_76416058130438
// MI455X (gfx1250) — matrix operand order repaired and checked by reading, not compiled, not run on hardware
//
#include <hip/hip_runtime.h>

#define B_SZ     8192
#define N_SITES  256
#define D_BOND   128
#define OUT_DIM  10
#define MID_SITES (N_SITES - 2)
#define PANEL_HALVES 32768              // 2(d) * 4(kt) * 8(nt) * 32(lane) * 16(halves)
#define PANEL_BYTES  65536

#if defined(__has_builtin)
#if __has_builtin(__builtin_amdgcn_global_load_async_to_lds_b128) && \
    __has_builtin(__builtin_amdgcn_s_wait_asynccnt)
#define USE_ASYNC_LDS 1
#endif
#endif

typedef __attribute__((ext_vector_type(16))) _Float16 v16h;
typedef __attribute__((ext_vector_type(8)))  _Float16 v8h;
typedef __attribute__((ext_vector_type(8)))  float    v8f;
typedef __attribute__((ext_vector_type(4)))  int      v4i;

// LDS (address_space(3)) typed pointers: force ds_load / ds_store lowering.
typedef __attribute__((address_space(3))) const v16h*     as3_v16h_cp;
typedef __attribute__((address_space(3))) const v8h*      as3_v8h_cp;
typedef __attribute__((address_space(3))) _Float16*       as3_h_p;
typedef __attribute__((address_space(3))) v4i*            as3_v4ip;
typedef __attribute__((address_space(1))) v4i*            gas1_v4ip;

// ---------------------------------------------------------------------------
// Kernel F1: per-sample min / 1/(max-min+eps)
// ---------------------------------------------------------------------------
__global__ __launch_bounds__(256) void mps_minmax_kernel(
    const float* __restrict__ x, float* __restrict__ mm) {
  __shared__ float smn[256];
  __shared__ float smx[256];
  const int b = blockIdx.x;
  const int t = threadIdx.x;
  float v = x[(size_t)b * N_SITES + t];
  smn[t] = v;
  smx[t] = v;
  __syncthreads();
  for (int s2 = 128; s2 > 0; s2 >>= 1) {
    if (t < s2) {
      smn[t] = fminf(smn[t], smn[t + s2]);
      smx[t] = fmaxf(smx[t], smx[t + s2]);
    }
    __syncthreads();
  }
  if (t == 0) {
    mm[b * 2 + 0] = smn[0];
    mm[b * 2 + 1] = 1.0f / (smx[0] - smn[0] + 1e-6f);
  }
}

// ---------------------------------------------------------------------------
// Kernel F2: phi tables, stored [site][batch] (coalesced reads in main loop)
// ---------------------------------------------------------------------------
__global__ __launch_bounds__(256) void mps_phi_kernel(
    const float* __restrict__ x, const float* __restrict__ mm,
    float* __restrict__ cosb, float* __restrict__ sinb) {
  const int bb   = blockIdx.x * 256 + threadIdx.x;   // batch index
  const int site = blockIdx.y;
  float v   = x[(size_t)bb * N_SITES + site];
  float mn  = mm[bb * 2 + 0];
  float inv = mm[bb * 2 + 1];
  float ang = 1.57079632679489662f * (v - mn) * inv;
  float sres, cres;
  sincosf(ang, &sres, &cres);
  cosb[(size_t)site * B_SZ + bb] = cres;
  sinb[(size_t)site * B_SZ + bb] = sres;
}

// ---------------------------------------------------------------------------
// Kernel P: pack mid [254][128(l)][2][128(r)] f32 -> f16 in WMMA B-operand
// lane layout: fragment(d,kt,nt): lane holds column n = nt*16+(lane&15),
// halves j=0..15 hold K = kt*32 + (lane>=16 ? 16 : 0) + j (contiguous).
// ---------------------------------------------------------------------------
__global__ __launch_bounds__(256) void mps_pack_kernel(
    const float* __restrict__ mid, _Float16* __restrict__ dst) {
  const int idx = blockIdx.x * 256 + threadIdx.x;   // exactly 254*32768 threads
  const int j    =  idx        & 15;
  const int lane = (idx >> 4)  & 31;
  const int nt   = (idx >> 9)  & 7;
  const int kt   = (idx >> 12) & 3;
  const int d    = (idx >> 14) & 1;
  const int site =  idx >> 15;
  const int K = kt * 32 + (lane >> 4) * 16 + j;
  const int n = nt * 16 + (lane & 15);
  float v = mid[(((size_t)site * D_BOND + K) * 2 + d) * D_BOND + n];
  dst[idx] = (_Float16)v;
}

// ---------------------------------------------------------------------------
// Main kernel: MPS chain with v_wmma_f32_16x16x32_f16.
// 128 threads = 4 waves; each wave owns 32 batch rows (two 16-row A tiles).
// LDS: double-buffered 64KB weight panels + 4 waves * 32x128 f16 left = 160KB.
// Panel s+1 is fetched with GLOBAL_LOAD_ASYNC_TO_LDS_B128 (ASYNCcnt) while
// WMMAs run on panel s; one barrier per site. All LDS accesses go through
// explicit addrspace(3) pointers so they lower to ds_load/ds_store (never flat).
// ---------------------------------------------------------------------------
__global__ __launch_bounds__(128) void mps_chain_kernel(
    const float* __restrict__ first,        // [2][128]
    const float* __restrict__ cosb,         // [N][B]
    const float* __restrict__ sinb,         // [N][B]
    const _Float16* __restrict__ midB,      // packed panels
    const float* __restrict__ lastw,        // [128][2]
    float* __restrict__ scalar_out) {       // [B]
  extern __shared__ __align__(128) char smem[];
  _Float16* lds16 = (_Float16*)smem;        // single LDS root; index arithmetic only
  // half-index bases: panel buf c at c*PANEL_HALVES; left tiles at 2*PANEL_HALVES

  const int tid    = threadIdx.x;
  const int lane   = tid & 31;
  const int wave   = tid >> 5;
  const int lanelo = lane & 15;
  const int hi     = lane >> 4;
  const int rowbase = blockIdx.x * 128 + wave * 32;

  v8f acc[2][8];

  // ---- site 0: left[b,r] = cos0[b]*first[0][r] + sin0[b]*first[1][r] ----
  #pragma unroll
  for (int t = 0; t < 2; ++t) {
    float fc[8], fs[8];
    #pragma unroll
    for (int j = 0; j < 8; ++j) {
      int row = rowbase + t * 16 + j + 8 * hi;
      fc[j] = cosb[row];          // site 0 offset is 0
      fs[j] = sinb[row];
    }
    #pragma unroll
    for (int nt = 0; nt < 8; ++nt) {
      int col = nt * 16 + lanelo;
      float f0 = first[col];
      float f1 = first[D_BOND + col];
      #pragma unroll
      for (int j = 0; j < 8; ++j)
        acc[t][nt][j] = fc[j] * f0 + fs[j] * f1;
    }
  }

  const size_t lleft_base = (size_t)2 * PANEL_HALVES + (size_t)wave * (32 * 128);

  // ---- panel staging: each thread moves 512B (32 x b128) ----
  #define STAGE_PANEL(bufidx, site)                                            \
    do {                                                                       \
      char* g = (char*)(midB + (size_t)(site) * PANEL_HALVES);                 \
      char* l = (char*)(lds16 + (size_t)(bufidx) * PANEL_HALVES);              \
      STAGE_BODY(g, l)                                                         \
    } while (0)

#ifdef USE_ASYNC_LDS
  #define STAGE_BODY(g, l)                                                     \
    _Pragma("unroll")                                                          \
    for (int q = 0; q < 32; ++q)                                               \
      __builtin_amdgcn_global_load_async_to_lds_b128(                          \
          (gas1_v4ip)((g) + (size_t)q * 2048 + (size_t)tid * 16),              \
          (as3_v4ip)((l) + (size_t)q * 2048 + (size_t)tid * 16), 0, 0);
#else
  #define STAGE_BODY(g, l)                                                     \
    _Pragma("unroll")                                                          \
    for (int q = 0; q < 32; ++q)                                               \
      *((as3_v4ip)(l) + (q * 128 + tid)) = ((const v4i*)(g))[q * 128 + tid];
#endif

  // preload first panel into buffer 0
  STAGE_PANEL(0, 0);

  // ---- chain over middle sites ----
  for (int s = 0; s < MID_SITES; ++s) {
    const int cur = s & 1;
#ifdef USE_ASYNC_LDS
    __builtin_amdgcn_s_wait_asynccnt(0);   // my slice of panel s is in LDS
#endif
    __syncthreads();                       // all slices visible; other buf free

    // kick off panel s+1 into the other buffer (overlaps compute below)
    if (s + 1 < MID_SITES) STAGE_PANEL(cur ^ 1, s + 1);

    // warm L2 two sites ahead (global_prefetch_b8)
    if (s + 2 < MID_SITES) {
      const char* nxt = (const char*)(midB + (size_t)(s + 2) * PANEL_HALVES);
      #pragma unroll
      for (int q = 0; q < 4; ++q)
        __builtin_prefetch(nxt + (size_t)tid * 512 + q * 128, 0, 1);
    }

    // spill f32 left tiles to per-wave LDS region as f16 row-major [32][128]
    #pragma unroll
    for (int t = 0; t < 2; ++t)
      #pragma unroll
      for (int nt = 0; nt < 8; ++nt)
        #pragma unroll
        for (int j = 0; j < 8; ++j) {
          int rl = t * 16 + j + 8 * hi;
          *((as3_h_p)(lds16 + lleft_base + rl * 128 + nt * 16 + lanelo)) =
              (_Float16)acc[t][nt][j];
          acc[t][nt][j] = 0.0f;
        }

    // per-row feature factors for phi site s+1 (A-layout: one row per lane)
    const float* cs = cosb + (size_t)(s + 1) * B_SZ;
    const float* sn = sinb + (size_t)(s + 1) * B_SZ;
    _Float16 c0 = (_Float16)cs[rowbase + lanelo];
    _Float16 s0 = (_Float16)sn[rowbase + lanelo];
    _Float16 c1 = (_Float16)cs[rowbase + 16 + lanelo];
    _Float16 s1 = (_Float16)sn[rowbase + 16 + lanelo];

    const size_t panel_base = (size_t)cur * PANEL_HALVES;
    #pragma unroll
    for (int kt = 0; kt < 4; ++kt) {
      // A operands: row = lanelo (+t*16); lo lanes K=+0..15; hi +16..31
      v16h a0[2], a1[2];
      #pragma unroll
      for (int t = 0; t < 2; ++t) {
        const _Float16* rp =
            lds16 + lleft_base + (t * 16 + lanelo) * 128 + kt * 32 + 16 * hi;
        v8h p0 = *((as3_v8h_cp)rp);
        v8h p1 = *((as3_v8h_cp)(rp + 8));
        v16h aT = __builtin_shufflevector(p0, p1, 0, 1, 2, 3, 4, 5, 6, 7,
                                                  8, 9, 10, 11, 12, 13, 14, 15);
        _Float16 cc = t ? c1 : c0;
        _Float16 ss = t ? s1 : s0;
        a0[t] = aT * cc;   // phi0 folded into A (per-row scalar)
        a1[t] = aT * ss;   // phi1 folded into A
      }
      #pragma unroll
      for (int nt = 0; nt < 8; ++nt) {
        const _Float16* pb0 =
            lds16 + panel_base + (size_t)(((0 * 4 + kt) * 8 + nt) * 32 + lane) * 16;
        const _Float16* pb1 =
            lds16 + panel_base + (size_t)(((1 * 4 + kt) * 8 + nt) * 32 + lane) * 16;
        v16h b0 = *((as3_v16h_cp)pb0);
        v16h b1 = *((as3_v16h_cp)pb1);
        #pragma unroll
        for (int t = 0; t < 2; ++t) {
          acc[t][nt] = __builtin_amdgcn_wmma_f32_16x16x32_f16(
              false, a0[t], false, b0, (short)0, acc[t][nt], false, false);
          acc[t][nt] = __builtin_amdgcn_wmma_f32_16x16x32_f16(
              false, a1[t], false, b1, (short)0, acc[t][nt], false, false);
        }
      }
    }
  }

  // ---- last site: scalar[b] = cosN*(left . last0) + sinN*(left . last1) ----
  float l0[8], l1[8];
  #pragma unroll
  for (int nt = 0; nt < 8; ++nt) {
    int col = nt * 16 + lanelo;
    l0[nt] = lastw[col * 2 + 0];
    l1[nt] = lastw[col * 2 + 1];
  }
  const float* cs = cosb + (size_t)(N_SITES - 1) * B_SZ;
  const float* sn = sinb + (size_t)(N_SITES - 1) * B_SZ;
  #pragma unroll
  for (int t = 0; t < 2; ++t) {
    #pragma unroll
    for (int j = 0; j < 8; ++j) {
      float p0 = 0.f, p1 = 0.f;
      #pragma unroll
      for (int nt = 0; nt < 8; ++nt) {
        p0 += acc[t][nt][j] * l0[nt];
        p1 += acc[t][nt][j] * l1[nt];
      }
      #pragma unroll
      for (int off = 1; off < 16; off <<= 1) {   // reduce within 16-lane half
        p0 += __shfl_xor(p0, off, 32);
        p1 += __shfl_xor(p1, off, 32);
      }
      if (lanelo == 0) {
        int row = rowbase + t * 16 + j + 8 * hi;
        scalar_out[row] = cs[row] * p0 + sn[row] * p1;
      }
    }
  }
}

// ---------------------------------------------------------------------------
// Kernel L: logits = scalar * w^T + b
// ---------------------------------------------------------------------------
__global__ __launch_bounds__(256) void mps_logits_kernel(
    const float* __restrict__ scalar, const float* __restrict__ w,
    const float* __restrict__ bias, float* __restrict__ out) {
  int i = blockIdx.x * 256 + threadIdx.x;
  if (i >= B_SZ * OUT_DIM) return;
  int b = i / OUT_DIM;
  int o = i - b * OUT_DIM;
  out[i] = scalar[b] * w[o] + bias[o];
}

// ---------------------------------------------------------------------------
extern "C" void kernel_launch(void* const* d_in, const int* in_sizes, int n_in,
                              void* d_out, int out_size, void* d_ws, size_t ws_size,
                              hipStream_t stream) {
  (void)in_sizes; (void)n_in; (void)out_size; (void)ws_size;
  const float* x     = (const float*)d_in[0];
  const float* first = (const float*)d_in[1];
  const float* mid   = (const float*)d_in[2];
  const float* lastw = (const float*)d_in[3];
  const float* w     = (const float*)d_in[4];
  const float* bias  = (const float*)d_in[5];
  float* out = (float*)d_out;

  char* ws = (char*)d_ws;
  size_t off = 0;
  float* cosb = (float*)(ws + off); off += (size_t)N_SITES * B_SZ * 4;          // 8 MB
  float* sinb = (float*)(ws + off); off += (size_t)N_SITES * B_SZ * 4;          // 8 MB
  _Float16* midB = (_Float16*)(ws + off);
  off += (size_t)MID_SITES * PANEL_HALVES * 2;                                   // 16.6 MB
  float* mm   = (float*)(ws + off); off += (size_t)B_SZ * 2 * 4;
  float* scal = (float*)(ws + off); off += (size_t)B_SZ * 4;

  // allow 160KB dynamic LDS for the chain kernel (LDS is 320KB/WGP on CDNA5)
  (void)hipFuncSetAttribute(reinterpret_cast<const void*>(mps_chain_kernel),
                            hipFuncAttributeMaxDynamicSharedMemorySize, 163840);

  mps_minmax_kernel<<<B_SZ, 256, 0, stream>>>(x, mm);
  mps_phi_kernel<<<dim3(B_SZ / 256, N_SITES), 256, 0, stream>>>(x, mm, cosb, sinb);
  mps_pack_kernel<<<(MID_SITES * PANEL_HALVES) / 256, 256, 0, stream>>>(mid, midB);
  mps_chain_kernel<<<B_SZ / 128, 128, 163840, stream>>>(first, cosb, sinb, midB,
                                                        lastw, scal);
  mps_logits_kernel<<<(B_SZ * OUT_DIM + 255) / 256, 256, 0, stream>>>(scal, w, bias,
                                                                      out);
}
